// Blipv2_83047487635715
// MI455X (gfx1250) — matrix operand order repaired and checked by reading, not compiled, not run on hardware
//
#include <hip/hip_runtime.h>
#include <hip/hip_bf16.h>

// ---------------- problem constants ----------------
#define BATCH   8
#define NTOK    4096
#define DIM     768
#define AGENTS  4
#define QKV3    2304          // 3*DIM
#define POOL    384           // DIM/2
#define BN      32768         // BATCH*NTOK
#define SCALEQ  0.03608439182435161f   // 768^-0.5

typedef __bf16 bf16;
typedef __attribute__((ext_vector_type(16))) __bf16 v16bf;
typedef __attribute__((ext_vector_type(8)))  __bf16 v8bf;
typedef __attribute__((ext_vector_type(4)))  __bf16 v4bf;
typedef __attribute__((ext_vector_type(8)))  float  v8f;

// ---------------- GEMM tile core (bf16 WMMA, wave32) ----------------
// Block: 256 threads = 8 waves (4 M x 2 N). Block tile 128x64, wave tile 32x32
// (2x2 v_wmma_f32_16x16x32_bf16 frags). Double-buffered LDS, 1 barrier/K-step.
#define TM 128
#define TN 64
#define APITCH 40   // halves; 80B row pitch -> conflict-spread, keeps 16B align
#define BPITCH 40

__device__ __forceinline__ void gemm_tile(
    const bf16* __restrict__ A, int lda,
    const bf16* __restrict__ Bm, int ldb,
    int K, int bM, int bN,
    bf16* As, bf16* Bs, v8f acc[2][2])
{
    const int tid  = threadIdx.x;
    const int lane = tid & 31;
    const int wave = tid >> 5;
    const int wm   = wave & 3;
    const int wn   = wave >> 2;
    const int arow = tid >> 2;            // 0..63
    const int aoff = (tid & 3) * 8;       // halves
    const size_t aBase0 = (size_t)(bM + arow)      * lda + aoff;
    const size_t aBase1 = (size_t)(bM + 64 + arow) * lda + aoff;
    const size_t bBase  = (size_t)(bN + arow)      * ldb + aoff;
    const int lm = lane & 15;
    const int aK = (lane < 16) ? 0 : 8;   // A frag K-chunk select
    const int bK = (lane < 16) ? 0 : 8;   // B frag K-chunk select

    __builtin_prefetch(Bm + bBase, 0, 1);   // global_prefetch_b8 on weights

    uint4 ra0 = *(const uint4*)(A  + aBase0);
    uint4 ra1 = *(const uint4*)(A  + aBase1);
    uint4 rb  = *(const uint4*)(Bm + bBase);
    *(uint4*)(As + arow * APITCH + aoff)        = ra0;
    *(uint4*)(As + (64 + arow) * APITCH + aoff) = ra1;
    *(uint4*)(Bs + arow * BPITCH + aoff)        = rb;
    __syncthreads();

    const int nk = K >> 5;
    for (int kt = 0; kt < nk; ++kt) {
        const int s = kt & 1;
        if (kt + 1 < nk) {
            const int kb = (kt + 1) << 5;
            ra0 = *(const uint4*)(A  + aBase0 + kb);
            ra1 = *(const uint4*)(A  + aBase1 + kb);
            rb  = *(const uint4*)(Bm + bBase  + kb);
        }
        const bf16* as = As + s * (TM * APITCH);
        const bf16* bs = Bs + s * (TN * BPITCH);
        v16bf a[2], b[2];
#pragma unroll
        for (int i = 0; i < 2; ++i) {
            const bf16* p = as + (wm * 32 + i * 16 + lm) * APITCH + aK;
            v8bf lo = *(const v8bf*)(p);
            v8bf hi = *(const v8bf*)(p + 16);
#pragma unroll
            for (int e = 0; e < 8; ++e) { a[i][e] = lo[e]; a[i][8 + e] = hi[e]; }
        }
#pragma unroll
        for (int j = 0; j < 2; ++j) {
            const bf16* p = bs + (wn * 32 + j * 16 + lm) * BPITCH + bK;
            v8bf lo = *(const v8bf*)(p);
            v8bf hi = *(const v8bf*)(p + 16);
#pragma unroll
            for (int e = 0; e < 8; ++e) { b[j][e] = lo[e]; b[j][8 + e] = hi[e]; }
        }
#pragma unroll
        for (int i = 0; i < 2; ++i)
#pragma unroll
            for (int j = 0; j < 2; ++j)
                acc[i][j] = __builtin_amdgcn_wmma_f32_16x16x32_bf16(
                    false, a[i], false, b[j], (short)0, acc[i][j], false, false);
        if (kt + 1 < nk) {
            const int s2 = s ^ 1;
            bf16* asw = As + s2 * (TM * APITCH);
            bf16* bsw = Bs + s2 * (TN * BPITCH);
            *(uint4*)(asw + arow * APITCH + aoff)        = ra0;
            *(uint4*)(asw + (64 + arow) * APITCH + aoff) = ra1;
            *(uint4*)(bsw + arow * BPITCH + aoff)        = rb;
        }
        __syncthreads();
    }
}

// qkv = x_bf @ Wqkv_bf^T  ->  bf16 (BN x 2304)
__global__ __launch_bounds__(256, 2)
void gemm_qkv_kernel(const bf16* __restrict__ A, const bf16* __restrict__ Bm,
                     bf16* __restrict__ C)
{
    __shared__ alignas(16) bf16 As[2 * TM * APITCH];
    __shared__ alignas(16) bf16 Bs[2 * TN * BPITCH];
    v8f acc[2][2] = {};
    const int bM = blockIdx.y * TM, bN = blockIdx.x * TN;
    gemm_tile(A, DIM, Bm, DIM, DIM, bM, bN, As, Bs, acc);
    const int lane = threadIdx.x & 31, wave = threadIdx.x >> 5;
    const int wm = wave & 3, wn = wave >> 2;
    const int lm = lane & 15, rhalf = (lane < 16) ? 0 : 8;
#pragma unroll
    for (int i = 0; i < 2; ++i)
#pragma unroll
        for (int j = 0; j < 2; ++j) {
            const int col = bN + wn * 32 + j * 16 + lm;
#pragma unroll
            for (int r = 0; r < 8; ++r) {
                const int row = bM + wm * 32 + i * 16 + rhalf + r;
                C[(size_t)row * QKV3 + col] = (bf16)acc[i][j][r];
            }
        }
}

// gelu(q @ Wp1^T + bp1) -> per-batch column-sum atomics into hbar (8 x 384).
// (mean over n commutes with the following Linear, so h1 is never materialized)
__global__ __launch_bounds__(256, 2)
void gemm_pool1_kernel(const bf16* __restrict__ A, const bf16* __restrict__ Bm,
                       const float* __restrict__ bp1, float* __restrict__ hbar)
{
    __shared__ alignas(16) bf16 As[2 * TM * APITCH];
    __shared__ alignas(16) bf16 Bs[2 * TN * BPITCH];
    v8f acc[2][2] = {};
    const int bM = blockIdx.y * TM, bN = blockIdx.x * TN;
    gemm_tile(A, QKV3, Bm, DIM, DIM, bM, bN, As, Bs, acc);  // q = cols 0..767 of qkv
    const int lane = threadIdx.x & 31, wave = threadIdx.x >> 5;
    const int wm = wave & 3, wn = wave >> 2;
    (void)wm;
    const int lm = lane & 15;
    const int batch = bM >> 12;   // 4096 rows per batch, TM=128 divides it
#pragma unroll
    for (int j = 0; j < 2; ++j) {
        const int col = bN + wn * 32 + j * 16 + lm;
        const float bias = bp1[col];
        float csum = 0.f;
#pragma unroll
        for (int i = 0; i < 2; ++i)
#pragma unroll
            for (int r = 0; r < 8; ++r) {
                const float v = acc[i][j][r] + bias;
                csum += 0.5f * v * (1.0f + erff(v * 0.70710678118654752f));
            }
        atomicAdd(&hbar[batch * POOL + col], csum);   // partial over this block's 128 rows
    }
}

// ---------------- small kernels ----------------
__global__ void zero_kernel(float* __restrict__ p, int n)
{
    const int i = blockIdx.x * 256 + threadIdx.x;
    if (i < n) p[i] = 0.f;
}

__global__ void cvt4_kernel(const float4* __restrict__ in, v4bf* __restrict__ out, int n4)
{
    const int i = blockIdx.x * 256 + threadIdx.x;
    if (i < n4) {
        const float4 v = in[i];
        v4bf o;
        o[0] = (bf16)v.x; o[1] = (bf16)v.y; o[2] = (bf16)v.z; o[3] = (bf16)v.w;
        out[i] = o;
    }
}

// at_raw[b, a*768+d] = (hbar[b]/4096) @ Wp2^T + bp2
__global__ void at_kernel(const float* __restrict__ hbar, const float* __restrict__ Wp2,
                          const float* __restrict__ bp2, float* __restrict__ at_raw)
{
    const int idx = blockIdx.x * 256 + threadIdx.x;      // 0..24575
    const int b = idx / (AGENTS * DIM);
    const int j = idx - b * (AGENTS * DIM);
    const float* h = hbar + b * POOL;
    const float* w = Wp2 + (size_t)j * POOL;
    float s = 0.f;
    for (int d = 0; d < POOL; ++d) s += h[d] * w[d];
    at_raw[idx] = s * (1.0f / (float)NTOK) + bp2[j];
}

// S[b,a,n] = SCALE * at_raw[b,a,:] . k[b,n,:]
__global__ __launch_bounds__(256)
void scores_kernel(const float* __restrict__ at_raw, const bf16* __restrict__ qkv,
                   float* __restrict__ S)
{
    __shared__ float atS[AGENTS * DIM];
    const int b = blockIdx.y;
    for (int i = threadIdx.x; i < AGENTS * DIM; i += 256) atS[i] = at_raw[b * AGENTS * DIM + i];
    __syncthreads();
    const int n = blockIdx.x * 256 + threadIdx.x;
    const bf16* krow = qkv + ((size_t)(b * NTOK + n)) * QKV3 + DIM;
    float a0 = 0, a1 = 0, a2 = 0, a3 = 0;
    for (int c = 0; c < DIM / 8; ++c) {
        v8bf kv = *(const v8bf*)(krow + c * 8);
#pragma unroll
        for (int e = 0; e < 8; ++e) {
            const float f = (float)kv[e];
            const int d = c * 8 + e;
            a0 += atS[d] * f;
            a1 += atS[DIM + d] * f;
            a2 += atS[2 * DIM + d] * f;
            a3 += atS[3 * DIM + d] * f;
        }
    }
    const size_t base = ((size_t)b * AGENTS) * NTOK + n;
    S[base]            = a0 * SCALEQ;
    S[base + NTOK]     = a1 * SCALEQ;
    S[base + 2 * NTOK] = a2 * SCALEQ;
    S[base + 3 * NTOK] = a3 * SCALEQ;
}

__device__ __forceinline__ float block_max(float v, float* red)
{
    const int tid = threadIdx.x;
    red[tid] = v; __syncthreads();
    for (int s = 128; s > 0; s >>= 1) {
        if (tid < s) red[tid] = fmaxf(red[tid], red[tid + s]);
        __syncthreads();
    }
    const float r = red[0]; __syncthreads(); return r;
}
__device__ __forceinline__ float block_sum(float v, float* red)
{
    const int tid = threadIdx.x;
    red[tid] = v; __syncthreads();
    for (int s = 128; s > 0; s >>= 1) {
        if (tid < s) red[tid] += red[tid + s];
        __syncthreads();
    }
    const float r = red[0]; __syncthreads(); return r;
}

// agent_rep = softmax_n(S)   (output #2)
__global__ __launch_bounds__(256)
void softmax_rep_kernel(const float* __restrict__ S, float* __restrict__ rep)
{
    __shared__ float red[256];
    const size_t row = blockIdx.x;
    const float* src = S + row * NTOK;
    float* dst = rep + row * NTOK;
    const int tid = threadIdx.x;
    float m = -3.4e38f;
    for (int i = tid; i < NTOK; i += 256) m = fmaxf(m, src[i]);
    m = block_max(m, red);
    float sum = 0.f;
    for (int i = tid; i < NTOK; i += 256) sum += expf(src[i] - m);
    sum = block_sum(sum, red);
    const float inv = 1.0f / sum;
    for (int i = tid; i < NTOK; i += 256) dst[i] = expf(src[i] - m) * inv;
}

// P = rep * (softmax_n(attn1+attn2) + 2) / 3   (H=1 => softmax over heads == 1)
__global__ __launch_bounds__(256)
void p_kernel(const float* __restrict__ a1, const float* __restrict__ a2,
              const float* __restrict__ rep, float* __restrict__ P)
{
    __shared__ float red[256];
    const size_t row = blockIdx.x;
    const float* s1 = a1 + row * NTOK;
    const float* s2 = a2 + row * NTOK;
    const float* rp = rep + row * NTOK;
    float* dst = P + row * NTOK;
    const int tid = threadIdx.x;
    float m = -3.4e38f;
    for (int i = tid; i < NTOK; i += 256) m = fmaxf(m, s1[i] + s2[i]);
    m = block_max(m, red);
    float sum = 0.f;
    for (int i = tid; i < NTOK; i += 256) sum += expf(s1[i] + s2[i] - m);
    sum = block_sum(sum, red);
    const float inv = 1.0f / sum;
    for (int i = tid; i < NTOK; i += 256)
        dst[i] = rp[i] * (expf(s1[i] + s2[i] - m) * inv + 2.0f) * (1.0f / 3.0f);
}

// at2 = at_raw * (softmax_d(tk1+tk2) + 2) / 3   (output #3, also used for q_attn)
__global__ __launch_bounds__(256)
void at2_kernel(const float* __restrict__ t1, const float* __restrict__ t2,
                const float* __restrict__ at_raw, float* __restrict__ at2_ws,
                float* __restrict__ at2_out)
{
    __shared__ float red[256];
    const size_t row = blockIdx.x;
    const float* p1 = t1 + row * DIM;
    const float* p2 = t2 + row * DIM;
    const float* ar = at_raw + row * DIM;
    const int tid = threadIdx.x;
    float m = -3.4e38f;
    for (int i = tid; i < DIM; i += 256) m = fmaxf(m, p1[i] + p2[i]);
    m = block_max(m, red);
    float sum = 0.f;
    for (int i = tid; i < DIM; i += 256) sum += expf(p1[i] + p2[i] - m);
    sum = block_sum(sum, red);
    const float inv = 1.0f / sum;
    for (int i = tid; i < DIM; i += 256) {
        const float v = ar[i] * (expf(p1[i] + p2[i] - m) * inv + 2.0f) * (1.0f / 3.0f);
        at2_ws[row * DIM + i] = v;
        at2_out[row * DIM + i] = v;
    }
}

// agent_v[b,a,d] += sum_n P[b,a,n] * v[b,n,d]   (split over n-chunks, f32 atomics)
__global__ __launch_bounds__(128)
void agentv_kernel(const float* __restrict__ P, const bf16* __restrict__ qkv,
                   float* __restrict__ agent_v)
{
    __shared__ float Pch[AGENTS][128];
    const int b = blockIdx.z;
    const int d = blockIdx.x * 128 + threadIdx.x;
    const int nbase = blockIdx.y * 512;
    float acc[AGENTS] = {0.f, 0.f, 0.f, 0.f};
    for (int c = 0; c < 4; ++c) {
        const int n0 = nbase + c * 128;
#pragma unroll
        for (int a = 0; a < AGENTS; ++a)
            Pch[a][threadIdx.x] = P[((size_t)(b * AGENTS + a)) * NTOK + n0 + threadIdx.x];
        __syncthreads();
        for (int nn = 0; nn < 128; ++nn) {
            const float vv = (float)qkv[((size_t)(b * NTOK + n0 + nn)) * QKV3 + 2 * DIM + d];
            acc[0] += Pch[0][nn] * vv;
            acc[1] += Pch[1][nn] * vv;
            acc[2] += Pch[2][nn] * vv;
            acc[3] += Pch[3][nn] * vv;
        }
        __syncthreads();
    }
#pragma unroll
    for (int a = 0; a < AGENTS; ++a)
        atomicAdd(&agent_v[((size_t)(b * AGENTS + a)) * DIM + d], acc[a]);
}

// gv[b,a,:] = agent_v[b,a,:] @ Wproj^T   (folds the 39-GFLOP proj GEMM away)
__global__ void gv_kernel(const float* __restrict__ agent_v, const float* __restrict__ Wproj,
                          float* __restrict__ gv)
{
    const int idx = blockIdx.x * 256 + threadIdx.x;   // 0..24575
    const int ba = idx / DIM;
    const int d = idx - ba * DIM;
    const float* av = agent_v + (size_t)ba * DIM;
    const float* w = Wproj + (size_t)d * DIM;
    float s = 0.f;
    for (int e = 0; e < DIM; ++e) s += av[e] * w[e];
    gv[idx] = s;
}

// out[b,n,:] = softmax_a(SCALE * q[b,n,:] . at2[b,a,:]) @ gv[b] + bproj
__global__ __launch_bounds__(256)
void final_kernel(const bf16* __restrict__ qkv, const float* __restrict__ at2,
                  const float* __restrict__ gv, const float* __restrict__ bproj,
                  float* __restrict__ out)
{
    __shared__ float at2S[AGENTS * DIM];
    __shared__ float gvS[AGENTS * DIM];
    __shared__ float bS[DIM];
    __shared__ float lS[64 * AGENTS];
    __shared__ float wS[64 * AGENTS];
    const int b = blockIdx.y;
    const int nbase = blockIdx.x * 64;
    const int tid = threadIdx.x;
    for (int i = tid; i < AGENTS * DIM; i += 256) {
        at2S[i] = at2[b * AGENTS * DIM + i];
        gvS[i] = gv[b * AGENTS * DIM + i];
    }
    for (int i = tid; i < DIM; i += 256) bS[i] = bproj[i];
    __syncthreads();
    {   // phase 1: 256 threads = 64 rows x 4 agents of q . at2 dots
        const int nl = tid >> 2, a = tid & 3;
        const bf16* qrow = qkv + ((size_t)(b * NTOK + nbase + nl)) * QKV3;
        const float* atr = at2S + a * DIM;
        float s = 0.f;
        for (int c = 0; c < DIM / 8; ++c) {
            v8bf qv = *(const v8bf*)(qrow + c * 8);
#pragma unroll
            for (int e = 0; e < 8; ++e) s += (float)qv[e] * atr[c * 8 + e];
        }
        lS[nl * AGENTS + a] = s * SCALEQ;
    }
    __syncthreads();
    if (tid < 64) {   // softmax over 4 agents
        const float l0 = lS[tid * 4 + 0], l1 = lS[tid * 4 + 1];
        const float l2 = lS[tid * 4 + 2], l3 = lS[tid * 4 + 3];
        const float m = fmaxf(fmaxf(l0, l1), fmaxf(l2, l3));
        const float e0 = expf(l0 - m), e1 = expf(l1 - m);
        const float e2 = expf(l2 - m), e3 = expf(l3 - m);
        const float inv = 1.0f / (e0 + e1 + e2 + e3);
        wS[tid * 4 + 0] = e0 * inv; wS[tid * 4 + 1] = e1 * inv;
        wS[tid * 4 + 2] = e2 * inv; wS[tid * 4 + 3] = e3 * inv;
    }
    __syncthreads();
    // phase 2: rank-4 expansion + bias, coalesced f32 stores
    for (int idx = tid; idx < 64 * DIM; idx += 256) {
        const int nl = idx / DIM;
        const int d = idx - nl * DIM;
        const float o = bS[d]
            + wS[nl * 4 + 0] * gvS[d]
            + wS[nl * 4 + 1] * gvS[DIM + d]
            + wS[nl * 4 + 2] * gvS[2 * DIM + d]
            + wS[nl * 4 + 3] * gvS[3 * DIM + d];
        out[((size_t)(b * NTOK + nbase + nl)) * DIM + d] = o;
    }
}

// ---------------- launch ----------------
extern "C" void kernel_launch(void* const* d_in, const int* in_sizes, int n_in,
                              void* d_out, int out_size, void* d_ws, size_t ws_size,
                              hipStream_t stream)
{
    const float* x     = (const float*)d_in[0];
    const float* attn1 = (const float*)d_in[1];
    const float* attn2 = (const float*)d_in[2];
    const float* tk1   = (const float*)d_in[3];
    const float* tk2   = (const float*)d_in[4];
    const float* Wqkv  = (const float*)d_in[5];
    const float* Wp1   = (const float*)d_in[6];
    const float* bp1   = (const float*)d_in[7];
    const float* Wp2   = (const float*)d_in[8];
    const float* bp2   = (const float*)d_in[9];
    const float* Wproj = (const float*)d_in[10];
    const float* bproj = (const float*)d_in[11];

    float* out = (float*)d_out;
    float* rep_out = out + (size_t)BN * DIM;                       // (8,1,4,4096)
    float* at2_out = rep_out + (size_t)BATCH * AGENTS * NTOK;      // (8,1,4,768)

    char* ws = (char*)d_ws;
    bf16*  x_bf    = (bf16*)ws;  ws += (size_t)BN * DIM * 2;
    bf16*  Wqkv_bf = (bf16*)ws;  ws += (size_t)QKV3 * DIM * 2;
    bf16*  Wp1_bf  = (bf16*)ws;  ws += (size_t)POOL * DIM * 2;
    bf16*  qkv_bf  = (bf16*)ws;  ws += (size_t)BN * QKV3 * 2;
    float* hbar    = (float*)ws; ws += (size_t)BATCH * POOL * 4;
    float* at_raw  = (float*)ws; ws += (size_t)BATCH * AGENTS * DIM * 4;
    float* Sl      = (float*)ws; ws += (size_t)BATCH * AGENTS * NTOK * 4;
    float* Pm      = (float*)ws; ws += (size_t)BATCH * AGENTS * NTOK * 4;
    float* agent_v = (float*)ws; ws += (size_t)BATCH * AGENTS * DIM * 4;
    float* at2_ws  = (float*)ws; ws += (size_t)BATCH * AGENTS * DIM * 4;
    float* gv      = (float*)ws; ws += (size_t)BATCH * AGENTS * DIM * 4;

    // zero the atomic accumulators (workspace is not re-zeroed between replays)
    zero_kernel<<<(BATCH * POOL + 255) / 256, 256, 0, stream>>>(hbar, BATCH * POOL);
    zero_kernel<<<(BATCH * AGENTS * DIM + 255) / 256, 256, 0, stream>>>(agent_v, BATCH * AGENTS * DIM);

    // fp32 -> bf16 conversions (halves GEMM-side HBM traffic)
    cvt4_kernel<<<((size_t)BN * DIM / 4 + 255) / 256, 256, 0, stream>>>(
        (const float4*)x, (v4bf*)x_bf, BN * DIM / 4);
    cvt4_kernel<<<((size_t)QKV3 * DIM / 4 + 255) / 256, 256, 0, stream>>>(
        (const float4*)Wqkv, (v4bf*)Wqkv_bf, QKV3 * DIM / 4);
    cvt4_kernel<<<((size_t)POOL * DIM / 4 + 255) / 256, 256, 0, stream>>>(
        (const float4*)Wp1, (v4bf*)Wp1_bf, POOL * DIM / 4);

    // qkv = x @ Wqkv^T  (116 GFLOP, bf16 WMMA)
    gemm_qkv_kernel<<<dim3(QKV3 / TN, BN / TM), 256, 0, stream>>>(x_bf, Wqkv_bf, qkv_bf);

    // hbar[b,:] = sum_n gelu(q @ Wp1^T + bp1)  (19.3 GFLOP, WMMA + atomic epilogue)
    gemm_pool1_kernel<<<dim3(POOL / TN, BN / TM), 256, 0, stream>>>(qkv_bf, Wp1_bf, bp1, hbar);

    // at = mean(h1) @ Wp2^T + bp2  (mean commuted before the Linear)
    at_kernel<<<BATCH * AGENTS * DIM / 256, 256, 0, stream>>>(hbar, Wp2, bp2, at_raw);

    // agent->token logits, softmax (agent_rep), bias fusion, agent_v
    scores_kernel<<<dim3(NTOK / 256, BATCH), 256, 0, stream>>>(at_raw, qkv_bf, Sl);
    softmax_rep_kernel<<<BATCH * AGENTS, 256, 0, stream>>>(Sl, rep_out);
    p_kernel<<<BATCH * AGENTS, 256, 0, stream>>>(attn1, attn2, rep_out, Pm);
    at2_kernel<<<BATCH * AGENTS, 256, 0, stream>>>(tk1, tk2, at_raw, at2_ws, at2_out);
    agentv_kernel<<<dim3(DIM / 128, NTOK / 512, BATCH), 128, 0, stream>>>(Pm, qkv_bf, agent_v);

    // fold proj through agent_v, then rank-4 expansion + softmax over agents
    gv_kernel<<<BATCH * AGENTS * DIM / 256, 256, 0, stream>>>(agent_v, Wproj, gv);
    final_kernel<<<dim3(NTOK / 64, BATCH), 256, 0, stream>>>(qkv_bf, at2_ws, gv, bproj, out);
}
